// CausalSelfAttention_19318762897672
// MI455X (gfx1250) — matrix operand order repaired and checked by reading, not compiled, not run on hardware
//
#include <hip/hip_runtime.h>

// ---------------------------------------------------------------------------
// CDNA5 (gfx1250, wave32) causal self-attention forward.
// Pipeline: f32->f16 conversions, QKV GEMM (WMMA f16), flash attention
// (WMMA f16 + online softmax), output projection GEMM (WMMA f16).
// All matmul fragments use the ISA 7.12.2 wave32 VGPR layouts directly.
// ---------------------------------------------------------------------------

typedef __attribute__((ext_vector_type(16))) _Float16 v16h;
typedef __attribute__((ext_vector_type(8)))  _Float16 v8h;
typedef __attribute__((ext_vector_type(8)))  float    v8f;

#define T_SEQ 4096
#define DM    1024
#define NH    16
#define DH    64

static __device__ __forceinline__ v8f wmma_f16(v16h a, v16h b, v8f c) {
  // D = A(16x32 f16) x B(32x16 f16) + C(16x16 f32)
  return __builtin_amdgcn_wmma_f32_16x16x32_f16(false, a, false, b, (short)0, c,
                                                false, false);
}

static __device__ __forceinline__ v8f zero8() {
  v8f z = {0.f, 0.f, 0.f, 0.f, 0.f, 0.f, 0.f, 0.f};
  return z;
}

// A-fragment (16x32 f16) from row-major [rows][ld]; rows r0..r0+15, k0..k0+31.
// ISA layout: lane L holds row L%16; halves[0..7]=K k0+8*(L/16)+0..7,
// halves[8..15]=K k0+8*(L/16)+16..23.
static __device__ __forceinline__ v16h load_frag_a(const _Float16* p, int ld,
                                                   int r0, int k0, int lane) {
  const int row = r0 + (lane & 15);
  const int kk  = k0 + ((lane & 16) >> 1);   // +8 for lanes 16..31
  const _Float16* q = p + (size_t)row * ld + kk;
  union { v16h v; v8h h[2]; } u;
  u.h[0] = *(const v8h*)(q);
  u.h[1] = *(const v8h*)(q + 16);
  return u.v;
}

// B-fragment (32x16 f16) with B[k][n] stored transposed: BT row-major [N][ld].
// ISA layout: lane L holds col n0+L%16; halves[0..7]=K k0+8*(L/16)+0..7,
// halves[8..15]=K k0+8*(L/16)+16..23.
static __device__ __forceinline__ v16h load_frag_b(const _Float16* p, int ld,
                                                   int n0, int k0, int lane) {
  const int row = n0 + (lane & 15);
  const int kk  = k0 + ((lane & 16) >> 1);   // +8 for lanes 16..31
  const _Float16* q = p + (size_t)row * ld + kk;
  union { v16h v; v8h h[2]; } u;
  u.h[0] = *(const v8h*)(q);
  u.h[1] = *(const v8h*)(q + 16);
  return u.v;
}

// --------------------------- pre-pass conversions --------------------------

__global__ void __launch_bounds__(256) cvt_f32_to_f16_kernel(
    const float* __restrict__ src, _Float16* __restrict__ dst, int n) {
  const int i = blockIdx.x * 256 + threadIdx.x;
  if (i < n) dst[i] = (_Float16)src[i];
}

// w [K][N] (f32, row-major) -> wT [N][K] (f16)
__global__ void __launch_bounds__(256) transpose_to_f16_kernel(
    const float* __restrict__ w, _Float16* __restrict__ wT, int K, int N) {
  const int i = blockIdx.x * 256 + threadIdx.x;
  if (i < K * N) {
    const int k = i / N;
    const int n = i - k * N;
    wT[(size_t)n * K + k] = (_Float16)w[i];
  }
}

// ------------------------------- QKV GEMM ----------------------------------
// C[4096,3072] = xh[4096,1024] @ w_qkv + b_qkv, scattered into per-head
// Q[h][t][d], K[h][t][d] and V^T[h][d][t] (all f16).

__global__ void __launch_bounds__(256) qkv_gemm_kernel(
    const _Float16* __restrict__ xh, const _Float16* __restrict__ wT,
    const float* __restrict__ bias, _Float16* __restrict__ Qh,
    _Float16* __restrict__ Kh, _Float16* __restrict__ VT) {
  const int lane = threadIdx.x & 31;
  const int wid  = blockIdx.x * 8 + (threadIdx.x >> 5);
  const int ntn  = (3 * DM) / 32;              // 96 tiles in N
  const int tm   = wid / ntn;
  const int tn   = wid - tm * ntn;
  const int m0   = tm * 32;
  const int n0   = tn * 32;

  v8f acc[2][2];
#pragma unroll
  for (int i = 0; i < 2; ++i)
#pragma unroll
    for (int j = 0; j < 2; ++j) acc[i][j] = zero8();

  for (int k0 = 0; k0 < DM; k0 += 32) {
    const v16h a0 = load_frag_a(xh, DM, m0,      k0, lane);
    const v16h a1 = load_frag_a(xh, DM, m0 + 16, k0, lane);
    const v16h b0 = load_frag_b(wT, DM, n0,      k0, lane);
    const v16h b1 = load_frag_b(wT, DM, n0 + 16, k0, lane);
    acc[0][0] = wmma_f16(a0, b0, acc[0][0]);
    acc[0][1] = wmma_f16(a0, b1, acc[0][1]);
    acc[1][0] = wmma_f16(a1, b0, acc[1][0]);
    acc[1][1] = wmma_f16(a1, b1, acc[1][1]);
  }

  const int nl    = lane & 15;
  const int half8 = (lane >> 4) << 3;
#pragma unroll
  for (int i = 0; i < 2; ++i) {
#pragma unroll
    for (int j = 0; j < 2; ++j) {
      const int n = n0 + 16 * j + nl;
      const float bb = bias[n];
#pragma unroll
      for (int v = 0; v < 8; ++v) {
        const int t = m0 + 16 * i + v + half8;
        const _Float16 hv = (_Float16)(acc[i][j][v] + bb);
        if (n < DM) {                       // Q
          const int hh = n >> 6, d = n & 63;
          Qh[((size_t)hh * T_SEQ + t) * DH + d] = hv;
        } else if (n < 2 * DM) {            // K
          const int nn = n - DM, hh = nn >> 6, d = nn & 63;
          Kh[((size_t)hh * T_SEQ + t) * DH + d] = hv;
        } else {                            // V, stored transposed [h][d][t]
          const int nn = n - 2 * DM, hh = nn >> 6, d = nn & 63;
          VT[((size_t)hh * DH + d) * T_SEQ + t] = hv;
        }
      }
    }
  }
}

// ---------------------------- flash attention ------------------------------
// One wave per (head, 16-row q tile). Streams keys in 32-wide steps:
// S = Q K^T (4 WMMAs), online softmax, P staged via LDS into an A-fragment,
// O += P V (4 WMMAs). Causal: only diagonal 16-blocks are masked; blocks
// above the diagonal are skipped.

__global__ void __launch_bounds__(256) attn_kernel(
    const _Float16* __restrict__ Qh, const _Float16* __restrict__ Kh,
    const _Float16* __restrict__ VTh, _Float16* __restrict__ Yh) {
  __shared__ _Float16 lds[8][16 * 32];     // 1 KB P-staging tile per wave

  const int lane  = threadIdx.x & 31;
  const int wave  = threadIdx.x >> 5;
  const int gwave = blockIdx.x * 8 + wave;
  const int qt    = gwave & 255;           // q tile (16 rows each)
  const int h     = gwave >> 8;            // head

  const _Float16* Q  = Qh  + (size_t)h * T_SEQ * DH;
  const _Float16* K  = Kh  + (size_t)h * T_SEQ * DH;
  const _Float16* VT = VTh + (size_t)h * DH * T_SEQ;

  // Q tile fragments (16 x 64 = two 16x32 A fragments), loaded once.
  const v16h qa0 = load_frag_a(Q, DH, qt * 16, 0,  lane);
  const v16h qa1 = load_frag_a(Q, DH, qt * 16, 32, lane);

  v8f o[4];
#pragma unroll
  for (int j = 0; j < 4; ++j) o[j] = zero8();

  float mrow[8], lrow[8];
#pragma unroll
  for (int v = 0; v < 8; ++v) { mrow[v] = -1e30f; lrow[v] = 0.f; }

  const float csc   = 0.125f * 1.44269504088896f;  // 1/sqrt(64) * log2(e)
  const int   nl    = lane & 15;
  const int   half8 = (lane >> 4) << 3;
  _Float16*   sp    = &lds[wave][0];

  for (int kb0 = 0; kb0 <= qt; kb0 += 2) {
    const int  kb1  = kb0 + 1;
    const bool has1 = (kb1 <= qt);         // wave-uniform

    v8f s0 = zero8();
    {
      const v16h kb00 = load_frag_b(K, DH, kb0 * 16, 0,  lane);
      const v16h kb01 = load_frag_b(K, DH, kb0 * 16, 32, lane);
      s0 = wmma_f16(qa0, kb00, s0);
      s0 = wmma_f16(qa1, kb01, s0);
    }
    v8f s1 = zero8();
    if (has1) {
      const v16h kb10 = load_frag_b(K, DH, kb1 * 16, 0,  lane);
      const v16h kb11 = load_frag_b(K, DH, kb1 * 16, 32, lane);
      s1 = wmma_f16(qa0, kb10, s1);
      s1 = wmma_f16(qa1, kb11, s1);
    }

    // Online softmax, per accumulator VGPR (row m = v + 8*(lane/16)).
#pragma unroll
    for (int v = 0; v < 8; ++v) {
      const int m = v + half8;
      float x0 = s0[v] * csc;
      if (kb0 == qt && nl > m) x0 = -1e30f;          // diagonal mask
      float x1 = has1 ? (s1[v] * csc) : -1e30f;
      if (has1 && kb1 == qt && nl > m) x1 = -1e30f;

      float bm = fmaxf(x0, x1);
      bm = fmaxf(bm, __shfl_xor(bm, 1, 32));
      bm = fmaxf(bm, __shfl_xor(bm, 2, 32));
      bm = fmaxf(bm, __shfl_xor(bm, 4, 32));
      bm = fmaxf(bm, __shfl_xor(bm, 8, 32));

      const float mn    = fmaxf(mrow[v], bm);
      const float alpha = exp2f(mrow[v] - mn);
      const float e0    = exp2f(x0 - mn);
      const float e1    = exp2f(x1 - mn);
      float rs = e0 + e1;
      rs += __shfl_xor(rs, 1, 32);
      rs += __shfl_xor(rs, 2, 32);
      rs += __shfl_xor(rs, 4, 32);
      rs += __shfl_xor(rs, 8, 32);
      lrow[v] = lrow[v] * alpha + rs;
      mrow[v] = mn;
#pragma unroll
      for (int j = 0; j < 4; ++j) o[j][v] *= alpha;

      // Stage P tile (f16) into LDS in row-major [m][32].
      sp[m * 32 + nl]      = (_Float16)e0;
      sp[m * 32 + 16 + nl] = (_Float16)e1;
    }

    // Re-read P as a 16x32 A fragment and do O += P @ V.
    const v16h pa = load_frag_a(sp, 32, 0, 0, lane);
#pragma unroll
    for (int j = 0; j < 4; ++j) {
      const v16h vb = load_frag_b(VT, T_SEQ, j * 16, kb0 * 16, lane);
      o[j] = wmma_f16(pa, vb, o[j]);
    }
  }

  // Normalize and store to attention output yh [T][DM] (f16).
#pragma unroll
  for (int v = 0; v < 8; ++v) {
    const int   t   = qt * 16 + v + half8;
    const float inv = 1.0f / lrow[v];
#pragma unroll
    for (int j = 0; j < 4; ++j) {
      Yh[(size_t)t * DM + h * DH + j * 16 + nl] = (_Float16)(o[j][v] * inv);
    }
  }
}

// --------------------------- output projection -----------------------------

__global__ void __launch_bounds__(256) outproj_gemm_kernel(
    const _Float16* __restrict__ yh, const _Float16* __restrict__ wT,
    const float* __restrict__ bias, float* __restrict__ out) {
  const int lane = threadIdx.x & 31;
  const int wid  = blockIdx.x * 8 + (threadIdx.x >> 5);
  const int ntn  = DM / 32;                // 32 tiles in N
  const int tm   = wid / ntn;
  const int tn   = wid - tm * ntn;
  const int m0   = tm * 32;
  const int n0   = tn * 32;

  v8f acc[2][2];
#pragma unroll
  for (int i = 0; i < 2; ++i)
#pragma unroll
    for (int j = 0; j < 2; ++j) acc[i][j] = zero8();

  for (int k0 = 0; k0 < DM; k0 += 32) {
    const v16h a0 = load_frag_a(yh, DM, m0,      k0, lane);
    const v16h a1 = load_frag_a(yh, DM, m0 + 16, k0, lane);
    const v16h b0 = load_frag_b(wT, DM, n0,      k0, lane);
    const v16h b1 = load_frag_b(wT, DM, n0 + 16, k0, lane);
    acc[0][0] = wmma_f16(a0, b0, acc[0][0]);
    acc[0][1] = wmma_f16(a0, b1, acc[0][1]);
    acc[1][0] = wmma_f16(a1, b0, acc[1][0]);
    acc[1][1] = wmma_f16(a1, b1, acc[1][1]);
  }

  const int nl    = lane & 15;
  const int half8 = (lane >> 4) << 3;
#pragma unroll
  for (int i = 0; i < 2; ++i) {
#pragma unroll
    for (int j = 0; j < 2; ++j) {
      const int n = n0 + 16 * j + nl;
      const float bb = bias[n];
#pragma unroll
      for (int v = 0; v < 8; ++v) {
        const int t = m0 + 16 * i + v + half8;
        out[(size_t)t * DM + n] = acc[i][j][v] + bb;
      }
    }
  }
}

// ------------------------------- launcher ----------------------------------

extern "C" void kernel_launch(void* const* d_in, const int* in_sizes, int n_in,
                              void* d_out, int out_size, void* d_ws,
                              size_t ws_size, hipStream_t stream) {
  const float* x     = (const float*)d_in[0];
  const float* w_qkv = (const float*)d_in[1];
  const float* b_qkv = (const float*)d_in[2];
  const float* w_out = (const float*)d_in[3];
  const float* b_out = (const float*)d_in[4];
  float*       out   = (float*)d_out;

  char* ws = (char*)d_ws;
  size_t off = 0;
  _Float16* xh    = (_Float16*)(ws + off); off += (size_t)T_SEQ * DM * 2;
  _Float16* wqkvT = (_Float16*)(ws + off); off += (size_t)3 * DM * DM * 2;
  _Float16* woutT = (_Float16*)(ws + off); off += (size_t)DM * DM * 2;
  _Float16* Qh    = (_Float16*)(ws + off); off += (size_t)T_SEQ * DM * 2;
  _Float16* Kh    = (_Float16*)(ws + off); off += (size_t)T_SEQ * DM * 2;
  _Float16* VT    = (_Float16*)(ws + off); off += (size_t)T_SEQ * DM * 2;
  _Float16* Yh    = (_Float16*)(ws + off); off += (size_t)T_SEQ * DM * 2;

  // 1) f32 -> f16 conversions / weight transposes.
  cvt_f32_to_f16_kernel<<<(T_SEQ * DM + 255) / 256, 256, 0, stream>>>(
      x, xh, T_SEQ * DM);
  transpose_to_f16_kernel<<<(DM * 3 * DM + 255) / 256, 256, 0, stream>>>(
      w_qkv, wqkvT, DM, 3 * DM);
  transpose_to_f16_kernel<<<(DM * DM + 255) / 256, 256, 0, stream>>>(
      w_out, woutT, DM, DM);

  // 2) QKV projection: 128 x 96 tiles of 32x32, 8 waves/block.
  qkv_gemm_kernel<<<(128 * 96) / 8, 256, 0, stream>>>(xh, wqkvT, b_qkv, Qh, Kh,
                                                      VT);

  // 3) Flash attention: 16 heads x 256 q-tiles = 4096 waves.
  attn_kernel<<<(NH * 256) / 8, 256, 0, stream>>>(Qh, Kh, VT, Yh);

  // 4) Output projection: 128 x 32 tiles of 32x32.
  outproj_gemm_kernel<<<(128 * 32) / 8, 256, 0, stream>>>(Yh, woutT, b_out,
                                                          out);
}
